// Agent_90503550861671
// MI455X (gfx1250) — compile-verified, matrix operand order checked by reading, not run on hardware
//
#include <hip/hip_runtime.h>
#include <hip/hip_bf16.h>
#include <math.h>

// ---------------- problem constants ----------------
#define T_STEPS 512
#define B_SZ    256
#define OBS_D   128
#define FEAT_D  512
#define H_D     256
#define ACT_D   16
#define N_TOT   (T_STEPS * B_SZ)   // 131072
#define G4      (4 * H_D)          // 1024
#define LOG2PI_F 1.8378770664093453f

// padded LDS strides (row-to-row offset == 4 mod 64 words -> conflict-free)
#define HBP 264    // bf16 h rows   (264 shorts = 132 words)
#define GP  1028   // gate rows     (1028 floats)
#define XTP 136    // bf16 x rows
#define H1P 520    // bf16 h1 rows
#define H2P 260    // f32 h2 rows

typedef __attribute__((ext_vector_type(16))) __bf16 v16bf;
typedef __attribute__((ext_vector_type(8)))  float  v8f;
typedef __attribute__((ext_vector_type(4)))  int    v4i_t;

union FragBF { v16bf v; unsigned u[8]; };
union FragF  { v8f   v; float    f[8]; };

// -------- CDNA5 async global->LDS path (guarded; sync fallback) --------
#if defined(__gfx1250__) && __has_builtin(__builtin_amdgcn_global_load_async_to_lds_b128)
  #define HAS_ASYNC_LDS 1
#else
  #define HAS_ASYNC_LDS 0
#endif
#if HAS_ASYNC_LDS
typedef __attribute__((address_space(1))) v4i_t GV4;
typedef __attribute__((address_space(3))) v4i_t LV4;
  #if __has_builtin(__builtin_amdgcn_s_wait_asynccnt)
    #define WAIT_ASYNC(n) __builtin_amdgcn_s_wait_asynccnt(n)
  #else
    #define WAIT_ASYNC(n) asm volatile("s_wait_asynccnt " #n)
  #endif
#else
  #define WAIT_ASYNC(n) do {} while (0)
#endif

__device__ __forceinline__ unsigned short f32_to_bf16(float x) {
  unsigned u = __float_as_uint(x);
  unsigned r = u + 0x7FFFu + ((u >> 16) & 1u);   // round-to-nearest-even
  return (unsigned short)(r >> 16);
}
#if __has_builtin(__builtin_amdgcn_cvt_pk_bf16_f32)
__device__ __forceinline__ unsigned pack2_bf16(float a, float b) {
  auto p = __builtin_amdgcn_cvt_pk_bf16_f32(a, b);
  unsigned u; __builtin_memcpy(&u, &p, 4); return u;
}
#else
__device__ __forceinline__ unsigned pack2_bf16(float a, float b) {
  return (unsigned)f32_to_bf16(a) | ((unsigned)f32_to_bf16(b) << 16);
}
#endif
__device__ __forceinline__ float sigmoidf(float x) { return 1.f / (1.f + expf(-x)); }

// Load one 16x32 bf16 WMMA fragment (A-layout; also valid as B when the
// source is the row-major weight W, since B[k][n] = W[n][k]).
// CDNA5: lane L(0-15): row=L, K=kb..kb+7 (v0..3) and K=16+kb.. (v4..7),
// kb = 0 for lanes 0-15, 8 for lanes 16-31.
__device__ __forceinline__ v16bf load_frag_b16(const unsigned short* tile, int ld, int k0) {
  int l  = threadIdx.x & 31;
  int r  = l & 15;
  int kb = (l >> 4) << 3;
  const unsigned* p0 = (const unsigned*)(tile + (size_t)r * ld + k0 + kb);
  const unsigned* p1 = (const unsigned*)(tile + (size_t)r * ld + k0 + 16 + kb);
  FragBF f;
#pragma unroll
  for (int i = 0; i < 4; ++i) { f.u[i] = p0[i]; f.u[4 + i] = p1[i]; }
  return f.v;
}

__device__ __forceinline__ v8f wmma_bf16(v16bf a, v16bf b, v8f c) {
  return __builtin_amdgcn_wmma_f32_16x16x32_bf16(false, a, false, b, (short)0, c, false, false);
}

// ---------------- helper kernels ----------------
__global__ void k_f32_to_bf16(const float* __restrict__ in,
                              unsigned short* __restrict__ out, int n) {
  int i = blockIdx.x * blockDim.x + threadIdx.x;
  if (i < n) out[i] = f32_to_bf16(in[i]);
}

__global__ void k_entropy(const float* __restrict__ logstd, float* __restrict__ out) {
  float e = 0.f;
#pragma unroll
  for (int j = 0; j < ACT_D; ++j) e += 0.5f + 0.5f * LOG2PI_F + logstd[j];
  int i = blockIdx.x * blockDim.x + threadIdx.x;
  if (i < N_TOT) out[i] = e;
}

// ---------------- fused MLP: relu(x@W1^T+b1) -> LN(.@W2^T+b2) -> relu -> bf16
__global__ __launch_bounds__(128)
void k_mlp(const float* __restrict__ x,
           const unsigned short* __restrict__ w1, const float* __restrict__ b1,
           const unsigned short* __restrict__ w2, const float* __restrict__ b2,
           const float* __restrict__ gw, const float* __restrict__ gb,
           unsigned short* __restrict__ feat) {
  __shared__ unsigned short xt[16][XTP];
  __shared__ unsigned short h1[16][H1P];
  __shared__ float          h2[16][H2P];
  __shared__ float          red[16][8][2];

  const int tid  = threadIdx.x;
  const int wave = tid >> 5;
  const int row0 = blockIdx.x * 16;

  { // stage x tile f32 -> bf16
    int m  = tid >> 3;
    int c0 = (tid & 7) * 16;
    const float* xr = x + (size_t)(row0 + m) * OBS_D + c0;
    unsigned* dst = (unsigned*)&xt[m][c0];
#pragma unroll
    for (int i = 0; i < 8; ++i) dst[i] = pack2_bf16(xr[2 * i], xr[2 * i + 1]);
  }
  __syncthreads();

  const int l    = tid & 31;
  const int n_in = l & 15;
  const int mh   = (l >> 4) << 3;

  { // GEMM1: [16,128] x W1^T -> relu -> h1(bf16)
    v16bf a[4];
#pragma unroll
    for (int kk = 0; kk < 4; ++kk) a[kk] = load_frag_b16(&xt[0][0], XTP, kk * 32);
    for (int nt = wave; nt < FEAT_D / 16; nt += 4) {
      int nb = nt * 16;
      FragF c;
      float bias = b1[nb + n_in];
#pragma unroll
      for (int r = 0; r < 8; ++r) c.f[r] = bias;
#pragma unroll
      for (int kk = 0; kk < 4; ++kk) {
        v16bf b = load_frag_b16(w1 + (size_t)nb * OBS_D, OBS_D, kk * 32);
        c.v = wmma_bf16(a[kk], b, c.v);
      }
#pragma unroll
      for (int r = 0; r < 8; ++r) {
        float v = c.f[r];
        h1[mh + r][nb + n_in] = f32_to_bf16(v > 0.f ? v : 0.f);
      }
    }
  }
  __syncthreads();

  { // GEMM2: [16,512] x W2^T -> +b2 -> h2(f32)
    for (int nt = wave; nt < H_D / 16; nt += 4) {
      int nb = nt * 16;
      FragF c;
      float bias = b2[nb + n_in];
#pragma unroll
      for (int r = 0; r < 8; ++r) c.f[r] = bias;
#pragma unroll
      for (int kk = 0; kk < FEAT_D / 32; ++kk) {
        v16bf a = load_frag_b16(&h1[0][0], H1P, kk * 32);
        v16bf b = load_frag_b16(w2 + (size_t)nb * FEAT_D, FEAT_D, kk * 32);
        c.v = wmma_bf16(a, b, c.v);
      }
#pragma unroll
      for (int r = 0; r < 8; ++r) h2[mh + r][nb + n_in] = c.f[r];
    }
  }
  __syncthreads();

  { // LayerNorm + ReLU -> feat (bf16); lane-consecutive LDS access
    int m = tid >> 3;
    int j = tid & 7;
    float s = 0.f, q = 0.f;
#pragma unroll
    for (int i = 0; i < 32; ++i) { float v = h2[m][j + 8 * i]; s += v; q += v * v; }
    red[m][j][0] = s; red[m][j][1] = q;
    __syncthreads();
    s = 0.f; q = 0.f;
#pragma unroll
    for (int i = 0; i < 8; ++i) { s += red[m][i][0]; q += red[m][i][1]; }
    float mean = s * (1.f / H_D);
    float var  = q * (1.f / H_D) - mean * mean;
    float rstd = rsqrtf(var + 1e-5f);
    unsigned short* fr = feat + (size_t)(row0 + m) * H_D;
#pragma unroll
    for (int i = 0; i < 32; ++i) {
      int k = j + 8 * i;
      float v = (h2[m][k] - mean) * rstd * gw[k] + gb[k];
      fr[k] = f32_to_bf16(v > 0.f ? v : 0.f);
    }
  }
}

// ---------------- gates_x = feat @ Wih^T + (bih + bhh)  -> f32 [N, 1024]
__global__ __launch_bounds__(256)
void k_gates(const unsigned short* __restrict__ feat,
             const unsigned short* __restrict__ wih,
             const float* __restrict__ bih, const float* __restrict__ bhh,
             float* __restrict__ gx) {
  const int tid  = threadIdx.x;
  const int wave = tid >> 5;
  const int row0 = blockIdx.x * 16;
  const unsigned short* at = feat + (size_t)row0 * H_D;

  v16bf a[8];
#pragma unroll
  for (int kk = 0; kk < 8; ++kk) a[kk] = load_frag_b16(at, H_D, kk * 32);

  const int l    = tid & 31;
  const int n_in = l & 15;
  const int mh   = (l >> 4) << 3;
  for (int nt = wave; nt < G4 / 16; nt += 8) {
    int nb = nt * 16;
    FragF c;
    float bias = bih[nb + n_in] + bhh[nb + n_in];
#pragma unroll
    for (int r = 0; r < 8; ++r) c.f[r] = bias;
#pragma unroll
    for (int kk = 0; kk < 8; ++kk) {
      v16bf b = load_frag_b16(wih + (size_t)nb * H_D, H_D, kk * 32);
      c.v = wmma_bf16(a[kk], b, c.v);
    }
    float* gr = gx + (size_t)row0 * G4 + nb + n_in;
#pragma unroll
    for (int r = 0; r < 8; ++r) gr[(size_t)(mh + r) * G4] = c.f[r];
  }
}

// ---------------- masked LSTM scan + fused heads ----------------
// dynamic-LDS layout (bytes)
#define SC_HS   0
#define SC_CS   16384
#define SC_HB   32768
#define SC_GB   41216
#define SC_GS0  107008
#define SC_GS1  172800
#define SC_RED  238592
#define SC_RED2 239616
#define SC_DST  240640
#define SCAN_LDS_BYTES 240704

struct ScanParams {
  const float* gx_a; const float* gx_c;
  const unsigned short* whh_a; const unsigned short* whh_c;
  const int* done;
  const float* h0_a; const float* c0_a;
  const float* h0_c; const float* c0_c;
  const float* lnw_a; const float* lnb_a;
  const float* lnw_c; const float* lnb_c;
  const float* mW; const float* mb; const float* logstd;
  const float* action;
  const float* hW; const float* hb;
  float* out;
};

__global__ __launch_bounds__(256)
void k_scan(ScanParams P) {
  extern __shared__ char smem[];
  float          (*hsld)[H_D] = (float (*)[H_D])(smem + SC_HS);
  float          (*csld)[H_D] = (float (*)[H_D])(smem + SC_CS);
  unsigned short (*hbld)[HBP] = (unsigned short (*)[HBP])(smem + SC_HB);
  float          (*gbuf)[GP]  = (float (*)[GP])(smem + SC_GB);
  float          (*gst0)[GP]  = (float (*)[GP])(smem + SC_GS0);
  float          (*gst1)[GP]  = (float (*)[GP])(smem + SC_GS1);
  float          (*red)[16]   = (float (*)[16])(smem + SC_RED);
  float          (*red2)[16]  = (float (*)[16])(smem + SC_RED2);
  float*         dstage       = (float*)(smem + SC_DST);

  const int tid    = threadIdx.x;
  const int wave   = tid >> 5;
  const int branch = blockIdx.x >> 4;            // 0 = actor, 1 = critic
  const int b0     = (blockIdx.x & 15) * 16;

  const float* gx           = branch ? P.gx_c  : P.gx_a;
  const unsigned short* whh = branch ? P.whh_c : P.whh_a;
  const float* h0  = branch ? P.h0_c  : P.h0_a;
  const float* c0  = branch ? P.c0_c  : P.c0_a;
  const float* lnw = branch ? P.lnw_c : P.lnw_a;
  const float* lnb = branch ? P.lnb_c : P.lnb_a;

  // stage one 16x1024 f32 gates tile into LDS (async on CDNA5)
  auto stage_tile = [&](int t, float (*dst)[GP]) {
    int m = tid >> 4, r16 = tid & 15;
    const char* grow = (const char*)(gx + ((size_t)t * B_SZ + b0 + m) * G4);
    char* lrow = (char*)(&dst[m][0]);
#pragma unroll
    for (int j = 0; j < 16; ++j) {
      int off = (r16 + 16 * j) * 16;
#if HAS_ASYNC_LDS
      __builtin_amdgcn_global_load_async_to_lds_b128(
          (GV4*)(grow + off), (LV4*)(lrow + off), 0, 0);
#else
      *reinterpret_cast<float4*>(lrow + off) =
          *reinterpret_cast<const float4*>(grow + off);
#endif
    }
  };

  for (int i = tid; i < 16 * H_D; i += 256) {
    int m = i >> 8, k = i & 255;
    float h = h0[(size_t)(b0 + m) * H_D + k];
    float c = c0[(size_t)(b0 + m) * H_D + k];
    hsld[m][k] = h; csld[m][k] = c; hbld[m][k] = f32_to_bf16(h);
  }
  stage_tile(0, gst0);

  const int l    = tid & 31;
  const int n_in = l & 15;
  const int mh   = (l >> 4) << 3;
  const int gm   = tid >> 4;   // head row group 0..15
  const int gj   = tid & 15;   // lane in group

  float ls = 0.f, inv_std = 1.f;
  if (branch == 0) { ls = P.logstd[gj]; inv_std = expf(-ls); }

  for (int t = 0; t < T_STEPS; ++t) {
    float (*gcur)[GP] = (t & 1) ? gst1 : gst0;
    float (*gnxt)[GP] = (t & 1) ? gst0 : gst1;
    __syncthreads();
    if (tid < 16) dstage[tid] = 1.f - (float)P.done[(size_t)t * B_SZ + b0 + tid];
    if (t + 1 < T_STEPS) { stage_tile(t + 1, gnxt); WAIT_ASYNC(16); }
    else { WAIT_ASYNC(0); }
    __syncthreads();

    // episode-reset mask
    for (int i = tid; i < 16 * H_D; i += 256) {
      int m = i >> 8, k = i & 255;
      float dm = dstage[m];
      float h = hsld[m][k] * dm;
      float c = csld[m][k] * dm;
      hsld[m][k] = h; csld[m][k] = c; hbld[m][k] = f32_to_bf16(h);
    }
    __syncthreads();

    // g = gates_x[t] + h @ Whh^T  ([16,256]x[256,1024] bf16 WMMA, f32 acc)
    {
      v16bf a[8];
#pragma unroll
      for (int kk = 0; kk < 8; ++kk) a[kk] = load_frag_b16(&hbld[0][0], HBP, kk * 32);
      for (int nt = wave; nt < G4 / 16; nt += 8) {
        int nb = nt * 16;
        if (nt + 8 < G4 / 16)
          __builtin_prefetch(whh + (size_t)(nb + 128 + (l << 2)) * H_D, 0, 1);
        FragF c;
#pragma unroll
        for (int r = 0; r < 8; ++r) c.f[r] = gcur[mh + r][nb + n_in];
#pragma unroll
        for (int kk = 0; kk < 8; ++kk) {
          v16bf b = load_frag_b16(whh + (size_t)nb * H_D, H_D, kk * 32);
          c.v = wmma_bf16(a[kk], b, c.v);
        }
#pragma unroll
        for (int r = 0; r < 8; ++r) gbuf[mh + r][nb + n_in] = c.f[r];
      }
    }
    __syncthreads();

    // gate nonlinearity (i,f,g,o); lane-consecutive LDS access
#pragma unroll
    for (int i = 0; i < 16; ++i) {
      int k = gj + 16 * i;
      float gi = gbuf[gm][k];
      float gf = gbuf[gm][H_D + k];
      float gg = gbuf[gm][2 * H_D + k];
      float go = gbuf[gm][3 * H_D + k];
      float c  = sigmoidf(gf) * csld[gm][k] + sigmoidf(gi) * tanhf(gg);
      float h  = sigmoidf(go) * tanhf(c);
      csld[gm][k] = c; hsld[gm][k] = h; hbld[gm][k] = f32_to_bf16(h);
    }
    __syncthreads();

    // fused head: LayerNorm(h) then policy logprob / value
    {
      float s = 0.f, q = 0.f;
#pragma unroll
      for (int i = 0; i < 16; ++i) { float v = hsld[gm][gj + 16 * i]; s += v; q += v * v; }
      red[gm][gj] = s; red2[gm][gj] = q;
      __syncthreads();
      s = 0.f; q = 0.f;
#pragma unroll
      for (int i = 0; i < 16; ++i) { s += red[gm][i]; q += red2[gm][i]; }
      float mean = s * (1.f / H_D);
      float var  = q * (1.f / H_D) - mean * mean;
      float rstd = rsqrtf(var + 1e-5f);
      size_t row = (size_t)t * B_SZ + b0 + gm;
      __syncthreads();
      if (branch == 0) {
        float acc = 0.f;
        const float* wrow = P.mW + (size_t)gj * H_D;
        for (int k = 0; k < H_D; ++k) {
          float v = (hsld[gm][k] - mean) * rstd * lnw[k] + lnb[k];
          acc += v * wrow[k];
        }
        float mu = acc + P.mb[gj];
        float z  = (P.action[row * ACT_D + gj] - mu) * inv_std;
        red[gm][gj] = -0.5f * z * z - ls - 0.5f * LOG2PI_F;
        __syncthreads();
        if (gj == 0) {
          float lp = 0.f;
#pragma unroll
          for (int i = 0; i < 16; ++i) lp += red[gm][i];
          P.out[row] = lp;
        }
      } else {
        float acc = 0.f;
#pragma unroll
        for (int i = 0; i < 16; ++i) {
          int k = gj + 16 * i;
          float v = (hsld[gm][k] - mean) * rstd * lnw[k] + lnb[k];
          acc += v * P.hW[k];
        }
        red[gm][gj] = acc;
        __syncthreads();
        if (gj == 0) {
          float val = P.hb[0];
#pragma unroll
          for (int i = 0; i < 16; ++i) val += red[gm][i];
          P.out[2 * (size_t)N_TOT + row] = val;
        }
      }
    }
  }

  __syncthreads();
  { // final hidden / cell states: ah, acell, ch, ccell
    size_t bh = 3 * (size_t)N_TOT + (size_t)branch * 2 * (size_t)(B_SZ * H_D);
    size_t bc = bh + (size_t)(B_SZ * H_D);
    for (int i = tid; i < 16 * H_D; i += 256) {
      int m = i >> 8, k = i & 255;
      P.out[bh + (size_t)(b0 + m) * H_D + k] = hsld[m][k];
      P.out[bc + (size_t)(b0 + m) * H_D + k] = csld[m][k];
    }
  }
}

// ---------------- host driver ----------------
extern "C" void kernel_launch(void* const* d_in, const int* in_sizes, int n_in,
                              void* d_out, int out_size, void* d_ws, size_t ws_size,
                              hipStream_t stream) {
  (void)in_sizes; (void)n_in; (void)out_size; (void)ws_size;
  const float* x      = (const float*)d_in[0];
  const int*   done   = (const int*)  d_in[1];
  const float* a_h0   = (const float*)d_in[2];
  const float* a_c0   = (const float*)d_in[3];
  const float* c_h0   = (const float*)d_in[4];
  const float* c_c0   = (const float*)d_in[5];
  const float* action = (const float*)d_in[6];
  const float* a_W1 = (const float*)d_in[7],  *a_b1 = (const float*)d_in[8];
  const float* a_W2 = (const float*)d_in[9],  *a_b2 = (const float*)d_in[10];
  const float* a_flnw = (const float*)d_in[11], *a_flnb = (const float*)d_in[12];
  const float* a_Wih = (const float*)d_in[13], *a_Whh = (const float*)d_in[14];
  const float* a_bih = (const float*)d_in[15], *a_bhh = (const float*)d_in[16];
  const float* a_lnw = (const float*)d_in[17], *a_lnb = (const float*)d_in[18];
  const float* c_W1 = (const float*)d_in[19], *c_b1 = (const float*)d_in[20];
  const float* c_W2 = (const float*)d_in[21], *c_b2 = (const float*)d_in[22];
  const float* c_flnw = (const float*)d_in[23], *c_flnb = (const float*)d_in[24];
  const float* c_Wih = (const float*)d_in[25], *c_Whh = (const float*)d_in[26];
  const float* c_bih = (const float*)d_in[27], *c_bhh = (const float*)d_in[28];
  const float* c_lnw = (const float*)d_in[29], *c_lnb = (const float*)d_in[30];
  const float* a_mW = (const float*)d_in[31], *a_mb = (const float*)d_in[32];
  const float* a_logstd = (const float*)d_in[33];
  const float* c_hW = (const float*)d_in[34], *c_hb = (const float*)d_in[35];
  float* out = (float*)d_out;

  char* ws = (char*)d_ws;
  size_t off = 0;
  float* gx_a = (float*)(ws + off); off += (size_t)N_TOT * G4 * sizeof(float);
  float* gx_c = (float*)(ws + off); off += (size_t)N_TOT * G4 * sizeof(float);
  unsigned short* feat_a = (unsigned short*)(ws + off); off += (size_t)N_TOT * H_D * 2;
  unsigned short* feat_c = (unsigned short*)(ws + off); off += (size_t)N_TOT * H_D * 2;
  unsigned short* w1a  = (unsigned short*)(ws + off); off += (size_t)FEAT_D * OBS_D * 2;
  unsigned short* w2a  = (unsigned short*)(ws + off); off += (size_t)H_D * FEAT_D * 2;
  unsigned short* wiha = (unsigned short*)(ws + off); off += (size_t)G4 * H_D * 2;
  unsigned short* whha = (unsigned short*)(ws + off); off += (size_t)G4 * H_D * 2;
  unsigned short* w1c  = (unsigned short*)(ws + off); off += (size_t)FEAT_D * OBS_D * 2;
  unsigned short* w2c  = (unsigned short*)(ws + off); off += (size_t)H_D * FEAT_D * 2;
  unsigned short* wihc = (unsigned short*)(ws + off); off += (size_t)G4 * H_D * 2;
  unsigned short* whhc = (unsigned short*)(ws + off); off += (size_t)G4 * H_D * 2;

  auto cvt = [&](const float* src, unsigned short* dst, int n) {
    k_f32_to_bf16<<<(n + 255) / 256, 256, 0, stream>>>(src, dst, n);
  };
  cvt(a_W1, w1a, FEAT_D * OBS_D);  cvt(a_W2, w2a, H_D * FEAT_D);
  cvt(a_Wih, wiha, G4 * H_D);      cvt(a_Whh, whha, G4 * H_D);
  cvt(c_W1, w1c, FEAT_D * OBS_D);  cvt(c_W2, w2c, H_D * FEAT_D);
  cvt(c_Wih, wihc, G4 * H_D);      cvt(c_Whh, whhc, G4 * H_D);

  k_entropy<<<N_TOT / 256, 256, 0, stream>>>(a_logstd, out + N_TOT);

  k_mlp<<<N_TOT / 16, 128, 0, stream>>>(x, w1a, a_b1, w2a, a_b2, a_flnw, a_flnb, feat_a);
  k_mlp<<<N_TOT / 16, 128, 0, stream>>>(x, w1c, c_b1, w2c, c_b2, c_flnw, c_flnb, feat_c);

  k_gates<<<N_TOT / 16, 256, 0, stream>>>(feat_a, wiha, a_bih, a_bhh, gx_a);
  k_gates<<<N_TOT / 16, 256, 0, stream>>>(feat_c, wihc, c_bih, c_bhh, gx_c);

  (void)hipFuncSetAttribute((const void*)k_scan,
                            hipFuncAttributeMaxDynamicSharedMemorySize,
                            SCAN_LDS_BYTES);
  ScanParams P;
  P.gx_a = gx_a; P.gx_c = gx_c; P.whh_a = whha; P.whh_c = whhc;
  P.done = done;
  P.h0_a = a_h0; P.c0_a = a_c0; P.h0_c = c_h0; P.c0_c = c_c0;
  P.lnw_a = a_lnw; P.lnb_a = a_lnb; P.lnw_c = c_lnw; P.lnb_c = c_lnb;
  P.mW = a_mW; P.mb = a_mb; P.logstd = a_logstd; P.action = action;
  P.hW = c_hW; P.hb = c_hb; P.out = out;
  k_scan<<<32, 256, SCAN_LDS_BYTES, stream>>>(P);
}
